// NTXentLoss_28621662060583
// MI455X (gfx1250) — matrix operand order repaired and checked by reading, not compiled, not run on hardware
//
#include <hip/hip_runtime.h>

// NT-Xent loss, fused normalized-GEMM + row logsumexp for MI455X (gfx1250).
// sim = zn zn^T / T computed as split-f16 WMMA (Ah*Bh + Ah*Bl + Al*Bh, f32 acc).
// B tiles double-buffered in LDS via CDNA5 async global->LDS (ASYNCcnt) path.

typedef __attribute__((ext_vector_type(16))) _Float16 v16h;
typedef __attribute__((ext_vector_type(8)))  _Float16 v8h;
typedef __attribute__((ext_vector_type(8)))  float    v8f;
typedef __attribute__((ext_vector_type(4)))  int      v4i;

#define BATCH 4096
#define NROW  8192
#define DIM   512
#define INV_T 2.0f      // 1 / TEMPERATURE
#define EPSN  1e-8f
#define NSLICE 8        // column slices (grid.x)
#define COLS_PER_SLICE (NROW / NSLICE)        // 1024
#define TILES_PER_SLICE (COLS_PER_SLICE / 16) // 64

// ---- CDNA5 async global->LDS support (guarded; falls back to sync stage) ----
#if defined(__has_builtin)
#  if __has_builtin(__builtin_amdgcn_global_load_async_to_lds_b128)
#    define HAVE_ASYNC_LDS 1
#  endif
#endif

#ifdef HAVE_ASYNC_LDS
typedef __attribute__((address_space(1))) v4i g_v4i;   // global (AS1)
typedef __attribute__((address_space(3))) v4i l_v4i;   // LDS (AS3)
__device__ __forceinline__ void async_cp16(const _Float16* g, _Float16* l) {
  __builtin_amdgcn_global_load_async_to_lds_b128((g_v4i*)g, (l_v4i*)l, 0, 0);
}
__device__ __forceinline__ void wait_async() {
#if __has_builtin(__builtin_amdgcn_s_wait_asynccnt)
  __builtin_amdgcn_s_wait_asynccnt(0);
#else
  asm volatile("s_wait_asynccnt 0x0" ::: "memory");
#endif
}
#else
__device__ __forceinline__ void async_cp16(const _Float16* g, _Float16* l) {
  *(v8h*)l = *(const v8h*)g;   // sync fallback: load->VGPR->ds_store
}
__device__ __forceinline__ void wait_async() {}
#endif

__device__ __forceinline__ v16h cat8(v8h a, v8h b) {
  return __builtin_shufflevector(a, b, 0,1,2,3,4,5,6,7,8,9,10,11,12,13,14,15);
}

// ---------------- Kernel 1: row norms + split-f16 conversion -----------------
// One wave per row (8 waves / 256-thread block). Each lane owns 16 elements.
__global__ __launch_bounds__(256) void nt_norm(const float* __restrict__ zi,
                                               const float* __restrict__ zj,
                                               _Float16* __restrict__ znh,
                                               _Float16* __restrict__ znl,
                                               float* __restrict__ inv) {
  const int row  = blockIdx.x * 8 + (threadIdx.x >> 5);
  const int lane = threadIdx.x & 31;
  const float* src = (row < BATCH) ? (zi + (size_t)row * DIM)
                                   : (zj + (size_t)(row - BATCH) * DIM);
  float vals[16];
  const float4* s4 = (const float4*)(src + lane * 16);
  float ss = 0.f;
#pragma unroll
  for (int q = 0; q < 4; ++q) {
    float4 f = s4[q];
    vals[q*4+0] = f.x; vals[q*4+1] = f.y; vals[q*4+2] = f.z; vals[q*4+3] = f.w;
    ss += f.x*f.x + f.y*f.y + f.z*f.z + f.w*f.w;
  }
#pragma unroll
  for (int msk = 16; msk >= 1; msk >>= 1) ss += __shfl_xor(ss, msk, 32);
  const float iv = 1.0f / fmaxf(sqrtf(ss), EPSN);

  _Float16* dh = znh + (size_t)row * DIM + lane * 16;
  _Float16* dl = znl + (size_t)row * DIM + lane * 16;
#pragma unroll
  for (int q = 0; q < 16; ++q) {
    const float zn = vals[q] * iv;
    const _Float16 h = (_Float16)zn;
    dh[q] = h;
    dl[q] = (_Float16)(zn - (float)h);
  }
  if (lane == 0) inv[row] = iv;
}

// ---------------- Kernel 2: positives (full fp32 dot) ------------------------
__global__ __launch_bounds__(256) void nt_pos(const float* __restrict__ zi,
                                              const float* __restrict__ zj,
                                              const float* __restrict__ inv,
                                              float* __restrict__ pos) {
  const int row  = blockIdx.x * 8 + (threadIdx.x >> 5);
  const int lane = threadIdx.x & 31;
  const int prow = (row < BATCH) ? row + BATCH : row - BATCH;
  const float* a = (row  < BATCH) ? (zi + (size_t)row  * DIM) : (zj + (size_t)(row  - BATCH) * DIM);
  const float* b = (prow < BATCH) ? (zi + (size_t)prow * DIM) : (zj + (size_t)(prow - BATCH) * DIM);
  const float4* a4 = (const float4*)(a + lane * 16);
  const float4* b4 = (const float4*)(b + lane * 16);
  float d = 0.f;
#pragma unroll
  for (int q = 0; q < 4; ++q) {
    float4 x = a4[q], y = b4[q];
    d += x.x*y.x + x.y*y.y + x.z*y.z + x.w*y.w;
  }
#pragma unroll
  for (int msk = 16; msk >= 1; msk >>= 1) d += __shfl_xor(d, msk, 32);
  if (lane == 0) pos[row] = d * inv[row] * inv[prow] * INV_T;
}

// ---------------- Kernel 3: fused WMMA GEMM + row sum(exp) -------------------
// Block = 128 threads (4 waves). Wave w owns rows [i0, i0+16); block sweeps one
// 1024-column slice. A frags (hi+lo) live in VGPRs; B tiles double-buffered in
// LDS via async global->LDS.
__global__ __launch_bounds__(128) void nt_sim(const _Float16* __restrict__ znh,
                                              const _Float16* __restrict__ znl,
                                              float* __restrict__ rowSumP) {
  constexpr int PITCH = DIM + 8;  // +16B pad per row: kill LDS bank conflicts
  __shared__ __align__(16) _Float16 Bh[2][16 * PITCH];
  __shared__ __align__(16) _Float16 Bl[2][16 * PITCH];

  const int lane = threadIdx.x & 31;
  const int wv   = threadIdx.x >> 5;     // 0..3
  const int m    = lane & 15;            // row-in-tile / col-in-tile
  const int h    = lane >> 4;            // half-wave select
  const int i0   = blockIdx.y * 64 + wv * 16;        // wave's global row base
  const int c0   = blockIdx.x * COLS_PER_SLICE;      // slice's column base

  const _Float16* arh = znh + (size_t)(i0 + m) * DIM;
  const _Float16* arl = znl + (size_t)(i0 + m) * DIM;

  // A-hi fragments: lane holds row (i0+m), K chunks at kb+8h and kb+16+8h.
  v16h Ah[16];
#pragma unroll
  for (int c = 0; c < 16; ++c) {
    const int kb = c * 32 + h * 8;
    Ah[c] = cat8(*(const v8h*)(arh + kb), *(const v8h*)(arh + kb + 16));
  }

  // Stage one 16-col B tile (hi+lo) into LDS buffer `buf`.
  auto stage = [&](int j0, int buf) {
    const int t = threadIdx.x;
#pragma unroll
    for (int q = 0; q < 8; ++q) {
      const int cid = t + q * 128;       // chunk id 0..1023 (8 halfs each)
      const int r   = cid >> 6;          // B row 0..15
      const int c8  = (cid & 63) * 8;    // half offset in row
      async_cp16(znh + (size_t)(j0 + r) * DIM + c8, &Bh[buf][r * PITCH + c8]);
      async_cp16(znl + (size_t)(j0 + r) * DIM + c8, &Bl[buf][r * PITCH + c8]);
    }
  };

  float racc[8] = {0.f, 0.f, 0.f, 0.f, 0.f, 0.f, 0.f, 0.f};

  // Prologue: fill buffer 0.
  stage(c0, 0);
  wait_async();
  __syncthreads();

  for (int jt = 0; jt < TILES_PER_SLICE; ++jt) {
    const int j0 = c0 + jt * 16;
    const int cb = jt & 1;
    // Prefetch next tile into the other buffer (its readers passed the last
    // barrier), overlapping with the WMMA work below.
    if (jt + 1 < TILES_PER_SLICE) stage(c0 + (jt + 1) * 16, cb ^ 1);

    v8f acc = {};
#pragma unroll
    for (int c = 0; c < 16; ++c) {
      // B frag: lane holds B-column (j0+m), K chunks at c*32 + 8h and c*32 + 16 + 8h.
      const _Float16* bhp = &Bh[cb][m * PITCH + c * 32 + h * 8];
      const _Float16* blp = &Bl[cb][m * PITCH + c * 32 + h * 8];
      const v16h bh = cat8(*(const v8h*)bhp, *(const v8h*)(bhp + 16));
      const v16h bl = cat8(*(const v8h*)blp, *(const v8h*)(blp + 16));
      // A-lo frag (loop-invariant: compiler keeps these register-resident).
      const int ka = c * 32 + h * 8;
      const v16h al = cat8(*(const v8h*)(arl + ka), *(const v8h*)(arl + ka + 16));
      acc = __builtin_amdgcn_wmma_f32_16x16x32_f16(false, al,    false, bh, (short)0, acc, false, false);
      acc = __builtin_amdgcn_wmma_f32_16x16x32_f16(false, Ah[c], false, bl, (short)0, acc, false, false);
      acc = __builtin_amdgcn_wmma_f32_16x16x32_f16(false, Ah[c], false, bh, (short)0, acc, false, false);
    }

    // C/D layout: VGPR v, lane -> (M = v + 8h, N = m). exp + diagonal mask.
    const int gj = j0 + m;
#pragma unroll
    for (int v = 0; v < 8; ++v) {
      const int gi = i0 + v + 8 * h;
      const float e = __expf(acc[v] * INV_T);
      racc[v] += (gi == gj) ? 0.f : e;
    }

    // Own prefetch done -> barrier: everyone's prefetch is LDS-visible, and
    // all reads of buffer cb are complete before it gets overwritten.
    wait_async();
    __syncthreads();
  }

  // Reduce across the 16 lanes that share each row; lanes m==0 write.
#pragma unroll
  for (int v = 0; v < 8; ++v) {
    float s = racc[v];
    s += __shfl_xor(s, 1, 32);
    s += __shfl_xor(s, 2, 32);
    s += __shfl_xor(s, 4, 32);
    s += __shfl_xor(s, 8, 32);
    if (m == 0)
      rowSumP[(size_t)blockIdx.x * NROW + (i0 + v + 8 * h)] = s;
  }
}

// ---------------- Kernel 4: final loss reduce --------------------------------
__global__ __launch_bounds__(256) void nt_loss(const float* __restrict__ rowSumP,
                                               const float* __restrict__ pos,
                                               float* __restrict__ out) {
  __shared__ float red[256];
  float s = 0.f;
  for (int i = threadIdx.x; i < NROW; i += 256) {
    float t = 0.f;
#pragma unroll
    for (int p = 0; p < NSLICE; ++p) t += rowSumP[(size_t)p * NROW + i];
    s += __logf(t) - pos[i];
  }
  red[threadIdx.x] = s;
  __syncthreads();
  for (int step = 128; step > 0; step >>= 1) {
    if (threadIdx.x < step) red[threadIdx.x] += red[threadIdx.x + step];
    __syncthreads();
  }
  if (threadIdx.x == 0) out[0] = red[0] / (float)NROW;
}

// ---------------- Launch -----------------------------------------------------
extern "C" void kernel_launch(void* const* d_in, const int* in_sizes, int n_in,
                              void* d_out, int out_size, void* d_ws, size_t ws_size,
                              hipStream_t stream) {
  (void)in_sizes; (void)n_in; (void)out_size; (void)ws_size;
  const float* zi = (const float*)d_in[0];
  const float* zj = (const float*)d_in[1];

  char* ws = (char*)d_ws;
  const size_t halfBytes = (size_t)NROW * DIM * sizeof(_Float16);  // 8 MiB
  _Float16* znh = (_Float16*)ws;
  _Float16* znl = (_Float16*)(ws + halfBytes);
  float* inv     = (float*)(ws + 2 * halfBytes);
  float* pos     = inv + NROW;
  float* rowSumP = pos + NROW;            // NSLICE * NROW floats

  nt_norm<<<NROW / 8, 256, 0, stream>>>(zi, zj, znh, znl, inv);
  nt_pos <<<NROW / 8, 256, 0, stream>>>(zi, zj, inv, pos);
  dim3 grid(NSLICE, NROW / 64);
  nt_sim <<<grid, 128, 0, stream>>>(znh, znl, rowSumP);
  nt_loss<<<1, 256, 0, stream>>>(rowSumP, pos, (float*)d_out);
}
